// Embedder_81466939670848
// MI455X (gfx1250) — matrix operand order repaired and checked by reading, not compiled, not run on hardware
//
#include <hip/hip_runtime.h>
#include <hip/hip_bf16.h>

typedef __attribute__((ext_vector_type(16))) __bf16 v16bf;
typedef __attribute__((ext_vector_type(8)))  float  v8f;

// ---------------- small elementwise / degree kernels ----------------

__global__ void k_set1(float* __restrict__ p, int n) {
    int i = blockIdx.x * blockDim.x + threadIdx.x;
    if (i < n) p[i] = 1.0f;   // self-loop contributes 1 to every node's degree
}

__global__ void k_zero(float* __restrict__ p, int n) {
    int i = blockIdx.x * blockDim.x + threadIdx.x;
    if (i < n) p[i] = 0.0f;
}

__global__ void k_deg_edges(const int* __restrict__ dst, float* __restrict__ deg, int E) {
    int i = blockIdx.x * blockDim.x + threadIdx.x;
    if (i < E) atomicAdd(&deg[dst[i]], 1.0f);
}

__global__ void k_rsqrt_inplace(float* __restrict__ p, int n) {
    int i = blockIdx.x * blockDim.x + threadIdx.x;
    if (i < n) p[i] = rsqrtf(p[i]);   // deg >= 1 always (self-loops)
}

template<int DOUT>
__global__ void k_fill_bias(float* __restrict__ out, const float* __restrict__ b, int N) {
    int i = blockIdx.x * blockDim.x + threadIdx.x;
    if (i < N * DOUT) out[i] = b[i & (DOUT - 1)];
}

// ---------------- weight pre-pack: f32 [DIN][DOUT] -> bf16 wave32 B-fragments ----------------
// Fragment order matches the GEMM's read:  idx = ((wv*(DIN/32) + kstep)*32 + lane)*16 + e
// where lane=(h<<4)|m holds column wv*16+m, K = kstep*32 + 8h + (e<8 ? e : e+8).

template<int DIN, int DOUT>
__global__ void k_pack_w(const float* __restrict__ W, __bf16* __restrict__ Wp) {
    int t = blockIdx.x * blockDim.x + threadIdx.x;
    constexpr int total = (DOUT / 16) * (DIN / 32) * 32;
    if (t >= total) return;
    const int lane = t & 31;
    const int rest = t >> 5;
    const int kkid = rest % (DIN / 32);
    const int wv   = rest / (DIN / 32);
    const int h = lane >> 4, m = lane & 15;
    const int kb = kkid * 32 + 8 * h;
    __bf16* o = Wp + (size_t)t * 16;
#pragma unroll
    for (int e = 0; e < 16; ++e)
        o[e] = (__bf16)W[(size_t)(kb + (e < 8 ? e : e + 8)) * DOUT + wv * 16 + m];
}

// ---------------- WMMA GEMM: XW = (relu?)X @ W,  bf16 inputs, f32 accum ----------------
// block = one 16-row tile; each wave32 = one 16-col tile of the output.
// B comes pre-packed (one contiguous 32B v16bf load per K-step).

template<int DIN, int DOUT, bool RELU>
__global__ void __launch_bounds__(32 * (DOUT / 16))
k_gemm(const float* __restrict__ X, const __bf16* __restrict__ Wp,
       float* __restrict__ XW, int nrows) {
    const int lane = threadIdx.x & 31;
    const int wv   = threadIdx.x >> 5;      // column tile index
    const int h    = lane >> 4;             // K-half selector
    const int m    = lane & 15;             // A row / B-C column within tile
    const int r0   = blockIdx.x * 16;

    int arow = r0 + m;
    if (arow >= nrows) arow = nrows - 1;    // clamp reads; keep EXEC full for WMMA
    const float* __restrict__ xrow = X + (size_t)arow * DIN;
    const __bf16* __restrict__ wbase =
        Wp + (((size_t)wv * (DIN / 32)) * 32 + lane) * 16;

    v8f acc = {};
#pragma unroll
    for (int kk = 0; kk < DIN; kk += 32) {
        // A (16x32 bf16): element e -> K = kk + (e<8 ? e : e+8) + 8h
        const int ka = kk + 8 * h;
        v16bf A;
#pragma unroll
        for (int e = 0; e < 8; ++e) {
            float a0 = xrow[ka + e];
            float a1 = xrow[ka + 16 + e];
            if (RELU) { a0 = fmaxf(a0, 0.0f); a1 = fmaxf(a1, 0.0f); }
            A[e]     = (__bf16)a0;
            A[e + 8] = (__bf16)a1;
        }
        // B (32x16 bf16): pre-packed fragment, contiguous 32 bytes per lane
        const v16bf B = *(const v16bf*)(wbase + (size_t)(kk >> 5) * 32 * 16);
        acc = __builtin_amdgcn_wmma_f32_16x16x32_bf16(
            /*neg_a=*/false, A, /*neg_b=*/false, B,
            /*c_mod=*/(short)0, acc, /*reuse_a=*/false, /*reuse_b=*/false);
    }
    // D (16x16 f32): VGPR r -> M = r + 8h, N = m
#pragma unroll
    for (int r = 0; r < 8; ++r) {
        int orow = r0 + r + 8 * h;
        if (orow < nrows)
            XW[(size_t)orow * DOUT + wv * 16 + m] = acc[r];
    }
}

// ---------------- edge gather + normalize + scatter-add (one wave32 / edge) ----------------

template<int DOUT>
__global__ void k_aggregate(const float* __restrict__ xw, const int* __restrict__ src,
                            const int* __restrict__ dst, const float* __restrict__ isd,
                            float* __restrict__ out, int E, int N) {
    int gw   = (blockIdx.x * blockDim.x + threadIdx.x) >> 5;
    int lane = threadIdx.x & 31;
    if (gw >= E + N) return;
    int s, d;
    if (gw < E) { s = src[gw]; d = dst[gw]; }
    else        { s = gw - E; d = s; }          // implicit self-loop edges
    float norm = isd[s] * isd[d];
    const float* __restrict__ row  = xw  + (size_t)s * DOUT;
    float*       __restrict__ orow = out + (size_t)d * DOUT;
    if constexpr (DOUT == 128) {
        float4 v = ((const float4*)row)[lane];
        int o = lane * 4;
        atomicAdd(&orow[o + 0], v.x * norm);
        atomicAdd(&orow[o + 1], v.y * norm);
        atomicAdd(&orow[o + 2], v.z * norm);
        atomicAdd(&orow[o + 3], v.w * norm);
    } else {  // DOUT == 64
        float2 v = ((const float2*)row)[lane];
        int o = lane * 2;
        atomicAdd(&orow[o + 0], v.x * norm);
        atomicAdd(&orow[o + 1], v.y * norm);
    }
}

// ---------------- global mean pool ----------------

__global__ void k_pool_sum(const float* __restrict__ hfeat, const int* __restrict__ batch,
                           float* __restrict__ pools, int N) {
    int i = blockIdx.x * blockDim.x + threadIdx.x;
    if (i < N * 64) {
        int node = i >> 6;
        atomicAdd(&pools[(size_t)batch[node] * 64 + (i & 63)], hfeat[i]);
    }
}

__global__ void k_pool_cnt(const int* __restrict__ batch, float* __restrict__ counts, int N) {
    int i = blockIdx.x * blockDim.x + threadIdx.x;
    if (i < N) atomicAdd(&counts[batch[i]], 1.0f);
}

__global__ void k_pool_div(const float* __restrict__ pools, const float* __restrict__ counts,
                           float* __restrict__ out, int G) {
    int i = blockIdx.x * blockDim.x + threadIdx.x;
    if (i < G * 64) out[i] = pools[i] / fmaxf(counts[i >> 6], 1.0f);
}

// ---------------- launch ----------------

extern "C" void kernel_launch(void* const* d_in, const int* in_sizes, int n_in,
                              void* d_out, int out_size, void* d_ws, size_t ws_size,
                              hipStream_t stream) {
    const float* x   = (const float*)d_in[0];   // [N,128]
    const int*   ei  = (const int*)  d_in[1];   // [2,E]
    const int*   bat = (const int*)  d_in[2];   // [N]
    const float* W1  = (const float*)d_in[3];
    const float* b1  = (const float*)d_in[4];
    const float* W2  = (const float*)d_in[5];
    const float* b2  = (const float*)d_in[6];
    const float* W3  = (const float*)d_in[7];
    const float* b3  = (const float*)d_in[8];

    const int N = in_sizes[0] / 128;
    const int E = in_sizes[1] / 2;
    const int G = out_size / 64;
    const int* srcp = ei;
    const int* dstp = ei + E;

    float* isd    = (float*)d_ws;                        // N
    float* buf0   = isd  + (((size_t)N + 63) & ~63ull);  // N*128 (GEMM out)
    float* buf1   = buf0 + (size_t)N * 128;              // N*128 (aggregate out / next in)
    float* pools  = buf1 + (size_t)N * 128;              // G*64
    float* counts = pools + (size_t)G * 64;              // G
    // packed bf16 weights (aligned to 64 floats = 256B)
    float* walign = counts + (((size_t)G + 63) & ~63ull);
    __bf16* Wp1 = (__bf16*)walign;                       // 128*128 bf16
    __bf16* Wp2 = Wp1 + 128 * 128;                       // 128*128 bf16
    __bf16* Wp3 = Wp2 + 128 * 128;                       // 128*64  bf16

    auto cdiv = [](long long a, long long b) { return (int)((a + b - 1) / b); };
    const int rowTiles  = cdiv(N, 16);
    const int aggBlocks = cdiv((long long)(E + N) * 32, 256);

    // pack weights into WMMA B-fragment layout (once per launch, L2-resident)
    k_pack_w<128, 128><<<cdiv(8 * 4 * 32, 256), 256, 0, stream>>>(W1, Wp1);
    k_pack_w<128, 128><<<cdiv(8 * 4 * 32, 256), 256, 0, stream>>>(W2, Wp2);
    k_pack_w<128,  64><<<cdiv(4 * 4 * 32, 256), 256, 0, stream>>>(W3, Wp3);

    // normalization coefficients
    k_set1         <<<cdiv(N, 256), 256, 0, stream>>>(isd, N);
    k_deg_edges    <<<cdiv(E, 256), 256, 0, stream>>>(dstp, isd, E);
    k_rsqrt_inplace<<<cdiv(N, 256), 256, 0, stream>>>(isd, N);

    // layer 1: xw = x@W1 ; agg = scatter(norm*gather(xw)) + b1
    k_gemm<128, 128, false><<<rowTiles, 256, 0, stream>>>(x, Wp1, buf0, N);
    k_fill_bias<128><<<cdiv((long long)N * 128, 256), 256, 0, stream>>>(buf1, b1, N);
    k_aggregate<128><<<aggBlocks, 256, 0, stream>>>(buf0, srcp, dstp, isd, buf1, E, N);

    // layer 2 (ReLU folded into the GEMM A-load)
    k_gemm<128, 128, true><<<rowTiles, 256, 0, stream>>>(buf1, Wp2, buf0, N);
    k_fill_bias<128><<<cdiv((long long)N * 128, 256), 256, 0, stream>>>(buf1, b2, N);
    k_aggregate<128><<<aggBlocks, 256, 0, stream>>>(buf0, srcp, dstp, isd, buf1, E, N);

    // layer 3 (Dout = 64)
    k_gemm<128, 64, true><<<rowTiles, 128, 0, stream>>>(buf1, Wp3, buf0, N);
    k_fill_bias<64><<<cdiv((long long)N * 64, 256), 256, 0, stream>>>(buf1, b3, N);
    k_aggregate<64><<<aggBlocks, 256, 0, stream>>>(buf0, srcp, dstp, isd, buf1, E, N);

    // global mean pool
    k_zero    <<<cdiv(G * 64 + G, 256), 256, 0, stream>>>(pools, G * 64 + G); // pools+counts contiguous
    k_pool_sum<<<cdiv((long long)N * 64, 256), 256, 0, stream>>>(buf1, bat, pools, N);
    k_pool_cnt<<<cdiv(N, 256), 256, 0, stream>>>(bat, counts, N);
    k_pool_div<<<cdiv(G * 64, 256), 256, 0, stream>>>(pools, counts, (float*)d_out, G);
}
